// VectorPoolAggregationModuleMSG_23364622090946
// MI455X (gfx1250) — matrix operand order repaired and checked by reading, not compiled, not run on hardware
//
#include <hip/hip_runtime.h>

typedef __attribute__((ext_vector_type(16))) _Float16 v16h;
typedef __attribute__((ext_vector_type(8)))  _Float16 v8h;
typedef __attribute__((ext_vector_type(8)))  float    v8f;

#define NPTS 16384
#define MQ   4096
#define CF   32
#define TV   27
#define KPAD 64          // 41 (=C+9) padded to 64 for two K=32 WMMA steps
#define F1   864         // TV*32
#define F2   128
#define TPTS 1024        // support points staged in LDS per tile (12 KB)
#define CUBE 4.0f        // MULT * R
#define SENT 1e38f

#if __has_builtin(__builtin_amdgcn_s_wait_asynccnt)
#define ASYNC_WAIT0() __builtin_amdgcn_s_wait_asynccnt(0)
#else
#define ASYNC_WAIT0() asm volatile("s_wait_asynccnt 0x0" ::: "memory")
#endif

__device__ __forceinline__ float grid_off(int i) {
  // jnp.linspace(-R + R/NV, R - R/NV, 3) with R=2, NV=3
  return (i == 0) ? (-4.0f / 3.0f) : ((i == 1) ? 0.0f : (4.0f / 3.0f));
}

// ---------------------------------------------------------------------------
// Kernel 1: 3-NN of each of the 27 grid centers among support points inside
// the Chebyshev cube (half-width 4) around the query. One block per query,
// 27 waves (wave32) = one wave per grid voxel. Support points are streamed
// through a double-buffered LDS tile with GLOBAL_LOAD_ASYNC_TO_LDS_B128
// (ASYNCcnt path): tile t+1 DMAs into LDS while the VALU processes tile t.
// ---------------------------------------------------------------------------
__global__ __launch_bounds__(TV * 32) void knn_kernel(
    const float* __restrict__ sup_xyz, const float* __restrict__ new_xyz,
    float* __restrict__ d3_out, int* __restrict__ idx_out,
    int* __restrict__ empty_out) {
  __shared__ float tile[2][TPTS * 3];   // 2 x 12 KB
  __shared__ float mD[TV * 96];
  __shared__ int   mI[TV * 96];
  __shared__ int   anyFound;

  const int tid  = threadIdx.x;
  const int m    = blockIdx.x;
  const int wv   = tid >> 5;  // 0..26 : grid voxel
  const int lane = tid & 31;

  const float qx = new_xyz[m * 3 + 0];
  const float qy = new_xyz[m * 3 + 1];
  const float qz = new_xyz[m * 3 + 2];
  const float gcx = qx + grid_off(wv / 9);
  const float gcy = qy + grid_off((wv / 3) % 3);
  const float gcz = qz + grid_off(wv % 3);

  if (tid == 0) anyFound = 0;

  // Each tile is TPTS*12 = 12288 bytes = 768 lanes x 16B.
  // Async DMA: per-lane 16B global -> LDS, no VGPR staging.
  auto issue_tile = [&](int buf, int p0) {
    if (tid < (TPTS * 12) / 16) {
      unsigned long long g =
          (unsigned long long)(uintptr_t)((const char*)sup_xyz +
                                          (size_t)p0 * 12 + (size_t)tid * 16);
      unsigned int l =
          (unsigned int)(uintptr_t)((char*)&tile[buf][0] + tid * 16);
      asm volatile("global_load_async_to_lds_b128 %0, %1, off"
                   :: "v"(l), "v"(g) : "memory");
    }
  };

  float b0 = SENT, b1 = SENT, b2 = SENT;  // lane-local top-3 (sorted)
  int   i0 = 0,    i1 = 0,    i2 = 0;
  bool  found = false;

  const int NT = NPTS / TPTS;             // 16 tiles
  issue_tile(0, 0);
  for (int t = 0; t < NT; ++t) {
    ASYNC_WAIT0();        // this wave's async writes for tile t landed
    __syncthreads();      // all waves' writes visible
    if (t + 1 < NT) issue_tile((t + 1) & 1, (t + 1) * TPTS);  // overlap

    const float* tb = &tile[t & 1][0];
    for (int j = lane; j < TPTS; j += 32) {
      const float px = tb[j * 3 + 0];
      const float py = tb[j * 3 + 1];
      const float pz = tb[j * 3 + 2];
      const float cheb =
          fmaxf(fabsf(px - qx), fmaxf(fabsf(py - qy), fabsf(pz - qz)));
      if (cheb <= CUBE) {
        found = true;
        const float dx = gcx - px, dy = gcy - py, dz = gcz - pz;
        const float dd = dx * dx + dy * dy + dz * dz;
        const int   gi = t * TPTS + j;
        if (dd < b2) {
          if (dd < b1) {
            b2 = b1; i2 = i1;
            if (dd < b0) { b1 = b0; i1 = i0; b0 = dd; i0 = gi; }
            else         { b1 = dd; i1 = gi; }
          } else { b2 = dd; i2 = gi; }
        }
      }
    }
    __syncthreads();      // all reads of this buffer done before it is reused
  }
  if (found) anyFound = 1;

  const int base = wv * 96 + lane * 3;
  mD[base + 0] = b0; mD[base + 1] = b1; mD[base + 2] = b2;
  mI[base + 0] = i0; mI[base + 1] = i1; mI[base + 2] = i2;
  __syncthreads();

  if (lane == 0) {  // merge 32 lane-local top-3 lists -> global top-3
    const int b = wv * 96;
    for (int s = 0; s < 3; ++s) {
      float best = 3.3e38f; int be = 0;
      for (int e = 0; e < 96; ++e) {
        const float v = mD[b + e];
        if (v < best) { best = v; be = e; }
      }
      mD[b + be] = 3.3e38f;
      float dO; int iO;
      if (best >= 1e37f) { dO = 1e10f; iO = 0; }   // <3 candidates: ref clamps inf->1e10
      else               { dO = best;  iO = mI[b + be]; }
      d3_out[(m * TV + wv) * 3 + s]  = dO;
      idx_out[(m * TV + wv) * 3 + s] = iO;
    }
  }
  if (tid == 0) empty_out[m] = (anyFound == 0) ? 1 : 0;
}

// ---------------------------------------------------------------------------
// Kernel 2: convert w1 (27,32,41) -> f16 padded (27,32,64); w2 (128,864) -> f16
// ---------------------------------------------------------------------------
__global__ void convert_weights(const float* __restrict__ w1,
                                const float* __restrict__ w2,
                                _Float16* __restrict__ w1h,
                                _Float16* __restrict__ w2h) {
  const int i = blockIdx.x * blockDim.x + threadIdx.x;
  const int T1 = TV * 32 * KPAD;       // 55296
  const int T2 = F2 * F1;              // 110592
  if (i < T1) {
    const int k  = i & (KPAD - 1);
    const int go = i >> 6;             // g*32 + o
    w1h[i] = (_Float16)((k < 41) ? w1[go * 41 + k] : 0.0f);
  } else if (i < T1 + T2) {
    const int j = i - T1;
    w2h[j] = (_Float16)w2[j];
  }
}

// ---------------------------------------------------------------------------
// Kernel 3: build feats (M,27,64) f16 = [IDW-interp features (32) |
// local xyz (9) | zero pad]. One thread per (m, voxel).
// ---------------------------------------------------------------------------
__global__ void feats_kernel(const float* __restrict__ d3,
                             const int* __restrict__ idx,
                             const int* __restrict__ empty,
                             const float* __restrict__ new_xyz,
                             const float* __restrict__ sup_xyz,
                             const float* __restrict__ sup_feat,
                             _Float16* __restrict__ feats) {
  const int t = blockIdx.x * blockDim.x + threadIdx.x;
  if (t >= MQ * TV) return;
  const int m = t / TV, g = t % TV;
  _Float16* out = feats + (size_t)t * KPAD;

  if (empty[m]) {
    #pragma unroll
    for (int j = 0; j < KPAD; ++j) out[j] = (_Float16)0.0f;
    return;
  }

  float wgt[3]; int id[3]; float norm = 0.0f;
  #pragma unroll
  for (int k = 0; k < 3; ++k) {
    const float d = d3[t * 3 + k];
    id[k]  = idx[t * 3 + k];
    wgt[k] = 1.0f / (d + 1e-8f);
    norm  += wgt[k];
  }
  const float inv = 1.0f / fmaxf(norm, 1e-8f);
  #pragma unroll
  for (int k = 0; k < 3; ++k) wgt[k] *= inv;

  float acc[CF];
  #pragma unroll
  for (int c = 0; c < CF; ++c) acc[c] = 0.0f;
  #pragma unroll
  for (int k = 0; k < 3; ++k) {
    const float* f = sup_feat + (size_t)id[k] * CF;
    const float  w = wgt[k];
    #pragma unroll
    for (int c = 0; c < CF; ++c) acc[c] += w * f[c];
  }
  #pragma unroll
  for (int c = 0; c < CF; ++c) out[c] = (_Float16)acc[c];

  const float gx = new_xyz[m * 3 + 0] + grid_off(g / 9);
  const float gy = new_xyz[m * 3 + 1] + grid_off((g / 3) % 3);
  const float gz = new_xyz[m * 3 + 2] + grid_off(g % 3);
  #pragma unroll
  for (int k = 0; k < 3; ++k) {
    const float* p = sup_xyz + (size_t)id[k] * 3;
    out[32 + k * 3 + 0] = (_Float16)(gx - p[0]);
    out[32 + k * 3 + 1] = (_Float16)(gy - p[1]);
    out[32 + k * 3 + 2] = (_Float16)(gz - p[2]);
  }
  #pragma unroll
  for (int j = 41; j < KPAD; ++j) out[j] = (_Float16)0.0f;
}

// ---------------------------------------------------------------------------
// WMMA fragment loaders per CDNA5 ISA §7.12.2 (wave32, f16 16x16x32):
//  A: lane holds row (lane&15); elements 0..7 = K[klo..klo+7],
//     elements 8..15 = K[klo+16..klo+23], klo = (lane>>4)*8.
//  B: lane holds column (lane&15); same K layout as A (klo = (lane>>4)*8).
// ---------------------------------------------------------------------------
__device__ __forceinline__ v16h load_frag_a(const _Float16* base) {
  const v8h lo = *(const v8h*)(base);
  const v8h hi = *(const v8h*)(base + 16);
  return __builtin_shufflevector(lo, hi, 0, 1, 2, 3, 4, 5, 6, 7,
                                 8, 9, 10, 11, 12, 13, 14, 15);
}
__device__ __forceinline__ v16h load_frag_b(const _Float16* base) {
  const v8h lo = *(const v8h*)(base);
  const v8h hi = *(const v8h*)(base + 16);
  return __builtin_shufflevector(lo, hi, 0, 1, 2, 3, 4, 5, 6, 7,
                                 8, 9, 10, 11, 12, 13, 14, 15);
}

// ---------------------------------------------------------------------------
// Kernel 4: grouped conv1d as WMMA: per (m-tile 16, voxel g):
//   y[m, g*32+o] = sum_k feats[m,g,k] * w1h[g,o,k]   (K=64, 2 steps of 32)
// Block = 64 threads (2 waves), wave w handles output half o in [w*16, w*16+16).
// ---------------------------------------------------------------------------
__global__ __launch_bounds__(64) void gemm1_kernel(
    const _Float16* __restrict__ feats, const _Float16* __restrict__ w1h,
    float* __restrict__ y) {
  const int m0   = blockIdx.x * 16;
  const int g    = blockIdx.y;
  const int wv   = threadIdx.x >> 5;   // n-half
  const int lane = threadIdx.x & 31;

  const int row = m0 + (lane & 15);
  const int klo = (lane >> 4) * 8;
  const _Float16* arow = feats + ((size_t)row * TV + g) * KPAD;

  const int ocol = wv * 16 + (lane & 15);
  const int kb   = (lane >> 4) * 8;
  const _Float16* brow = w1h + ((size_t)g * 32 + ocol) * KPAD;

  v8f acc = {};
  #pragma unroll
  for (int ks = 0; ks < 2; ++ks) {
    const v16h a = load_frag_a(arow + ks * 32 + klo);
    const v16h b = load_frag_b(brow + ks * 32 + kb);
    acc = __builtin_amdgcn_wmma_f32_16x16x32_f16(false, a, false, b,
                                                 (short)0, acc, false, false);
  }
  // D layout: VGPR v -> row m0 + v + 8*(lane>>4), col = lane&15
  const int mrow   = m0 + 8 * (lane >> 4);
  const int colOut = g * 32 + ocol;
  #pragma unroll
  for (int v = 0; v < 8; ++v)
    y[(size_t)(mrow + v) * F1 + colOut] = acc[v];
}

// ---------------------------------------------------------------------------
// Kernel 5: per-column mean/var (biased, ddof=0) over rows. Block per column.
// ---------------------------------------------------------------------------
__global__ __launch_bounds__(256) void col_stats_kernel(
    const float* __restrict__ x, int rows, int cols,
    float* __restrict__ mean, float* __restrict__ var) {
  __shared__ float s1[256], s2[256];
  const int c = blockIdx.x;
  float s = 0.0f, q = 0.0f;
  for (int r = threadIdx.x; r < rows; r += 256) {
    const float v = x[(size_t)r * cols + c];
    s += v; q += v * v;
  }
  s1[threadIdx.x] = s; s2[threadIdx.x] = q;
  __syncthreads();
  for (int st = 128; st > 0; st >>= 1) {
    if (threadIdx.x < st) {
      s1[threadIdx.x] += s1[threadIdx.x + st];
      s2[threadIdx.x] += s2[threadIdx.x + st];
    }
    __syncthreads();
  }
  if (threadIdx.x == 0) {
    const float mu = s1[0] / (float)rows;
    mean[c] = mu;
    var[c]  = s2[0] / (float)rows - mu * mu;
  }
}

// ---------------------------------------------------------------------------
// Kernel 6: BN + ReLU -> f16 (feeds GEMM2)
// ---------------------------------------------------------------------------
__global__ void bn_relu_f16_kernel(const float* __restrict__ x,
                                   const float* __restrict__ mean,
                                   const float* __restrict__ var,
                                   const float* __restrict__ gamma,
                                   const float* __restrict__ beta,
                                   _Float16* __restrict__ out,
                                   int cols, int total) {
  const int i = blockIdx.x * blockDim.x + threadIdx.x;
  if (i >= total) return;
  const int c = i % cols;
  const float v = (x[i] - mean[c]) * rsqrtf(var[c] + 1e-5f) * gamma[c] + beta[c];
  out[i] = (_Float16)fmaxf(v, 0.0f);
}

// ---------------------------------------------------------------------------
// Kernel 7: z = y_h (4096x864 f16) @ w2h^T (864x128) with WMMA, f32 accum.
// Block = 256 threads (8 waves), wave w -> output cols [w*16, w*16+16).
// ---------------------------------------------------------------------------
__global__ __launch_bounds__(256) void gemm2_kernel(
    const _Float16* __restrict__ yh, const _Float16* __restrict__ w2h,
    float* __restrict__ z) {
  const int m0   = blockIdx.x * 16;
  const int wv   = threadIdx.x >> 5;   // 0..7
  const int lane = threadIdx.x & 31;

  const int row = m0 + (lane & 15);
  const int klo = (lane >> 4) * 8;
  const _Float16* arow = yh + (size_t)row * F1;

  const int n  = wv * 16 + (lane & 15);
  const int kb = (lane >> 4) * 8;
  const _Float16* brow = w2h + (size_t)n * F1;

  v8f acc = {};
  for (int ks = 0; ks < F1 / 32; ++ks) {   // 27 K-steps
    const v16h a = load_frag_a(arow + ks * 32 + klo);
    const v16h b = load_frag_b(brow + ks * 32 + kb);
    acc = __builtin_amdgcn_wmma_f32_16x16x32_f16(false, a, false, b,
                                                 (short)0, acc, false, false);
  }
  const int mrow = m0 + 8 * (lane >> 4);
  #pragma unroll
  for (int v = 0; v < 8; ++v)
    z[(size_t)(mrow + v) * F2 + wv * 16 + (lane & 15)] = acc[v];
}

// ---------------------------------------------------------------------------
// Kernel 8: BN + ReLU -> f32 final output
// ---------------------------------------------------------------------------
__global__ void bn_relu_f32_kernel(const float* __restrict__ x,
                                   const float* __restrict__ mean,
                                   const float* __restrict__ var,
                                   const float* __restrict__ gamma,
                                   const float* __restrict__ beta,
                                   float* __restrict__ out,
                                   int cols, int total) {
  const int i = blockIdx.x * blockDim.x + threadIdx.x;
  if (i >= total) return;
  const int c = i % cols;
  const float v = (x[i] - mean[c]) * rsqrtf(var[c] + 1e-5f) * gamma[c] + beta[c];
  out[i] = fmaxf(v, 0.0f);
}

// ---------------------------------------------------------------------------
extern "C" void kernel_launch(void* const* d_in, const int* in_sizes, int n_in,
                              void* d_out, int out_size, void* d_ws, size_t ws_size,
                              hipStream_t stream) {
  const float* sup_xyz  = (const float*)d_in[0];  // (16384,3)
  const float* sup_feat = (const float*)d_in[1];  // (16384,32)
  const float* new_xyz  = (const float*)d_in[2];  // (4096,3)
  const float* w1       = (const float*)d_in[3];  // (27,32,41)
  const float* bn1_g    = (const float*)d_in[4];  // (864)
  const float* bn1_b    = (const float*)d_in[5];
  const float* w2       = (const float*)d_in[6];  // (128,864)
  const float* bn2_g    = (const float*)d_in[7];  // (128)
  const float* bn2_b    = (const float*)d_in[8];
  float* out = (float*)d_out;                     // (4096,128)

  // workspace carve-out (256B aligned), total ~40.6 MB
  char* wp = (char*)d_ws;
  auto carve = [&](size_t bytes) -> void* {
    void* p = (void*)wp;
    wp += (bytes + 255) & ~(size_t)255;
    return p;
  };
  float*     d3    = (float*)    carve((size_t)MQ * TV * 3 * 4);
  int*       idx   = (int*)      carve((size_t)MQ * TV * 3 * 4);
  int*       empty = (int*)      carve((size_t)MQ * 4);
  _Float16*  feats = (_Float16*) carve((size_t)MQ * TV * KPAD * 2);
  _Float16*  w1h   = (_Float16*) carve((size_t)TV * 32 * KPAD * 2);
  _Float16*  w2h   = (_Float16*) carve((size_t)F2 * F1 * 2);
  float*     y     = (float*)    carve((size_t)MQ * F1 * 4);
  _Float16*  yh    = (_Float16*) carve((size_t)MQ * F1 * 2);
  float*     z     = (float*)    carve((size_t)MQ * F2 * 4);
  float*     mean1 = (float*)    carve(F1 * 4);
  float*     var1  = (float*)    carve(F1 * 4);
  float*     mean2 = (float*)    carve(F2 * 4);
  float*     var2  = (float*)    carve(F2 * 4);

  // 1) 3-NN search: block per query, 27 waves = one per grid voxel,
  //    async double-buffered LDS tiles
  knn_kernel<<<MQ, TV * 32, 0, stream>>>(sup_xyz, new_xyz, d3, idx, empty);

  // 2) weight conversion to f16 (padded)
  {
    const int tot = TV * 32 * KPAD + F2 * F1;
    convert_weights<<<(tot + 255) / 256, 256, 0, stream>>>(w1, w2, w1h, w2h);
  }

  // 3) feature assembly: interp + local xyz, f16 padded K
  feats_kernel<<<(MQ * TV + 255) / 256, 256, 0, stream>>>(
      d3, idx, empty, new_xyz, sup_xyz, sup_feat, feats);

  // 4) grouped conv via WMMA
  gemm1_kernel<<<dim3(MQ / 16, TV), 64, 0, stream>>>(feats, w1h, y);

  // 5) BN1 stats + apply(+ReLU) -> f16
  col_stats_kernel<<<F1, 256, 0, stream>>>(y, MQ, F1, mean1, var1);
  bn_relu_f16_kernel<<<(MQ * F1 + 255) / 256, 256, 0, stream>>>(
      y, mean1, var1, bn1_g, bn1_b, yh, F1, MQ * F1);

  // 6) dense GEMM via WMMA
  gemm2_kernel<<<MQ / 16, 256, 0, stream>>>(yh, w2h, z);

  // 7) BN2 stats + apply(+ReLU) -> output
  col_stats_kernel<<<F2, 256, 0, stream>>>(z, MQ, F2, mean2, var2);
  bn_relu_f32_kernel<<<(MQ * F2 + 255) / 256, 256, 0, stream>>>(
      z, mean2, var2, bn2_g, bn2_b, out, F2, MQ * F2);
}
